// BantamMoEBlock_29652454211595
// MI455X (gfx1250) — matrix operand order repaired and checked by reading, not compiled, not run on hardware
//
#include <hip/hip_runtime.h>
#include <hip/hip_bf16.h>
#include <math.h>

// ---------------- problem constants ----------------
constexpr int BB = 8, TT = 2048, HH = 1024, FF = 2048, EE = 8;
constexpr int SS = BB * TT;          // 16384 tokens
constexpr int CAP = 2560;            // ceil(1.25 * S / E)
constexpr float AUX_W = 0.01f, Z_W = 0.001f;

typedef __attribute__((ext_vector_type(16))) __bf16 v16bf;
typedef __attribute__((ext_vector_type(8)))  __bf16 v8bf;
typedef __attribute__((ext_vector_type(8)))  float  v8f;

#define WMMA_BF16(a, b, c) \
  __builtin_amdgcn_wmma_f32_16x16x32_bf16(false, (a), false, (b), (short)0, (c), false, false)

// ---------------- WMMA tile loaders (per ISA 7.12.2, wave32) ----------------
// A (16x32 bf16, MxK): lane 0-15 -> M=lane, K in {8*kh..8*kh+7, 16+8*kh..}, kh=lane>>4
__device__ __forceinline__ v16bf load_a_tile(const __bf16* base, int ld, int m0, int k0, int lane) {
  const __bf16* p = base + (size_t)(m0 + (lane & 15)) * ld + k0 + ((lane >> 4) << 3);
  v8bf lo = *(const v8bf*)p;          // K = 8*kh + 0..7
  v8bf hi = *(const v8bf*)(p + 16);   // K = 16 + 8*kh + 0..7
  return __builtin_shufflevector(lo, hi, 0,1,2,3,4,5,6,7,8,9,10,11,12,13,14,15);
}
// B (32x16 bf16, KxN), weights stored transposed [N][K]: lane -> N=lane&15, K in {8*kh..8*kh+7, 16+8*kh..}, kh=lane>>4
__device__ __forceinline__ v16bf load_b_tile(const __bf16* baseT, int ld, int n0, int k0, int lane) {
  const __bf16* p = baseT + (size_t)(n0 + (lane & 15)) * ld + k0 + ((lane >> 4) << 3);
  v8bf lo = *(const v8bf*)p;
  v8bf hi = *(const v8bf*)(p + 16);
  return __builtin_shufflevector(lo, hi, 0,1,2,3,4,5,6,7,8,9,10,11,12,13,14,15);
}

// ---------------- 0) zero output ----------------
__global__ void moe_zero_kernel(float* out, size_t n) {
  size_t i = (size_t)blockIdx.x * blockDim.x + threadIdx.x;
  size_t st = (size_t)gridDim.x * blockDim.x;
  for (; i < n; i += st) out[i] = 0.0f;
}

// ---------------- 1) transpose fp32 [R][C] -> bf16 [C][R], per expert ----------------
__global__ __launch_bounds__(256) void moe_transpose_kernel(const float* __restrict__ src,
                                                            __bf16* __restrict__ dst,
                                                            int R, int C) {
  __shared__ float ts[32][33];
  int e = blockIdx.z;
  const float* s = src + (size_t)e * R * C;
  __bf16* d = dst + (size_t)e * R * C;
  int c0 = blockIdx.x * 32, r0 = blockIdx.y * 32;
  for (int i = threadIdx.y; i < 32; i += 8)
    ts[i][threadIdx.x] = s[(size_t)(r0 + i) * C + c0 + threadIdx.x];
  __syncthreads();
  for (int i = threadIdx.y; i < 32; i += 8)
    d[(size_t)(c0 + i) * R + r0 + threadIdx.x] = (__bf16)ts[threadIdx.x][i];
}

// ---------------- 2) router: one wave per token ----------------
__global__ __launch_bounds__(256) void moe_router_kernel(const float* __restrict__ X,
                                                         const float* __restrict__ rw,
                                                         const float* __restrict__ rb,
                                                         int* __restrict__ eidx,
                                                         float* __restrict__ gate,
                                                         float* __restrict__ partials) {
  int wave = threadIdx.x >> 5, lane = threadIdx.x & 31;
  int t = blockIdx.x * 8 + wave;
  float acc[EE];
#pragma unroll
  for (int e = 0; e < EE; ++e) acc[e] = 0.0f;
  const float* xr = X + (size_t)t * HH;
  for (int h = lane; h < HH; h += 32) {
    float xv = xr[h];
    const float* w = rw + (size_t)h * EE;
#pragma unroll
    for (int e = 0; e < EE; ++e) acc[e] += xv * w[e];
  }
#pragma unroll
  for (int e = 0; e < EE; ++e)
    for (int o = 16; o > 0; o >>= 1) acc[e] += __shfl_xor(acc[e], o, 32);

  __shared__ float part[8][EE + 1];
  if (lane == 0) {
    float l[EE];
#pragma unroll
    for (int e = 0; e < EE; ++e) l[e] = acc[e] + rb[e];
    float mx = l[0]; int arg = 0;
#pragma unroll
    for (int e = 1; e < EE; ++e) if (l[e] > mx) { mx = l[e]; arg = e; }
    float p[EE], se = 0.0f;
#pragma unroll
    for (int e = 0; e < EE; ++e) { p[e] = __expf(l[e] - mx); se += p[e]; }
    float inv = 1.0f / se;
#pragma unroll
    for (int e = 0; e < EE; ++e) p[e] *= inv;
    float z = mx + __logf(se);
    float top = p[arg];
    eidx[t] = arg;
    gate[t] = top / (top + 1e-9f);
#pragma unroll
    for (int e = 0; e < EE; ++e) part[wave][e] = p[e];
    part[wave][EE] = z * z;
  }
  __syncthreads();
  if (threadIdx.x < EE + 1) {
    float s = 0.0f;
#pragma unroll
    for (int w = 0; w < 8; ++w) s += part[w][threadIdx.x];
    partials[(size_t)blockIdx.x * (EE + 1) + threadIdx.x] = s;
  }
}

// ---------------- 3) capacity scan (single block, token order) ----------------
__global__ __launch_bounds__(256) void moe_scan_kernel(const int* __restrict__ eidx,
                                                       int* __restrict__ slot,
                                                       int* __restrict__ count,
                                                       int* __restrict__ rowlimit) {
  __shared__ int s_base[EE];
  __shared__ int s_wcnt[8][EE];
  __shared__ int s_woff[8][EE];
  int wave = threadIdx.x >> 5, lane = threadIdx.x & 31;
  if (threadIdx.x < EE) s_base[threadIdx.x] = 0;
  __syncthreads();
  for (int chunk = 0; chunk < SS / 256; ++chunk) {
    int t = chunk * 256 + threadIdx.x;
    int e = eidx[t];
    unsigned my_mask = 0;
#pragma unroll
    for (int ex = 0; ex < EE; ++ex) {
      unsigned m = (unsigned)__ballot(e == ex);
      if (ex == e) my_mask = m;
      if (lane == ex) s_wcnt[wave][ex] = __popc(m);
    }
    int rank = __popc(my_mask & ((1u << lane) - 1u));
    __syncthreads();
    if (wave == 0 && lane < EE) {
      int ex = lane, run = s_base[ex];
#pragma unroll
      for (int w = 0; w < 8; ++w) { s_woff[w][ex] = run; run += s_wcnt[w][ex]; }
      s_base[ex] = run;
    }
    __syncthreads();
    slot[t] = s_woff[wave][e] + rank;
    __syncthreads();
  }
  if (threadIdx.x < EE) {
    int c = s_base[threadIdx.x];
    count[threadIdx.x] = c;
    rowlimit[threadIdx.x] = c < CAP ? c : CAP;
  }
}

// ---------------- 4) gather kept tokens -> bf16 dispatch buffer ----------------
__global__ __launch_bounds__(256) void moe_gather_kernel(const float* __restrict__ X,
                                                         const int* __restrict__ eidx,
                                                         const int* __restrict__ slot,
                                                         __bf16* __restrict__ Xd,
                                                         int* __restrict__ tok) {
  int wave = threadIdx.x >> 5, lane = threadIdx.x & 31;
  int t = blockIdx.x * 8 + wave;
  int e = eidx[t], s = slot[t];
  if (s >= CAP) return;
  size_t row = (size_t)e * CAP + s;
  if (lane == 0) tok[row] = t;
  const float* xr = X + (size_t)t * HH;
  __bf16* dr = Xd + row * HH;
  for (int i = lane; i < HH; i += 32) dr[i] = (__bf16)xr[i];
}

// ---------------- 5) GEMM1 fused: h = silu(Xd@w1+b1) * (Xd@w2+b2) ----------------
// Block tile 128(M) x 64(N), 8 waves as 4x2, 32x32 per wave (2x2 subtiles x 2 paths).
__global__ __launch_bounds__(256) void moe_gemm1_kernel(const __bf16* __restrict__ Xd,
                                                        const __bf16* __restrict__ w1t,
                                                        const __bf16* __restrict__ w2t,
                                                        const float* __restrict__ b1,
                                                        const float* __restrict__ b2,
                                                        __bf16* __restrict__ hbuf,
                                                        const int* __restrict__ rowlimit) {
  int e = blockIdx.z;
  int rl = rowlimit[e];
  if ((int)blockIdx.y * 128 >= rl) return;
  int wave = threadIdx.x >> 5, lane = threadIdx.x & 31;
  int wm = wave >> 1, wn = wave & 1;
  int M0 = blockIdx.y * 128 + wm * 32;
  int N0 = blockIdx.x * 64 + wn * 32;
  const __bf16* A  = Xd  + (size_t)e * CAP * HH;
  const __bf16* B1 = w1t + (size_t)e * FF * HH;
  const __bf16* B2 = w2t + (size_t)e * FF * HH;
  v8f acc1[2][2], acc2[2][2];   // [mi][ni]
#pragma unroll
  for (int mi = 0; mi < 2; ++mi)
#pragma unroll
    for (int ni = 0; ni < 2; ++ni) {
      acc1[mi][ni] = {0.f,0.f,0.f,0.f,0.f,0.f,0.f,0.f};
      acc2[mi][ni] = {0.f,0.f,0.f,0.f,0.f,0.f,0.f,0.f};
    }
  for (int k0 = 0; k0 < HH; k0 += 32) {
    v16bf a0 = load_a_tile(A, HH, M0,      k0, lane);
    v16bf a1 = load_a_tile(A, HH, M0 + 16, k0, lane);
    v16bf p0 = load_b_tile(B1, HH, N0,      k0, lane);
    v16bf p1 = load_b_tile(B1, HH, N0 + 16, k0, lane);
    v16bf q0 = load_b_tile(B2, HH, N0,      k0, lane);
    v16bf q1 = load_b_tile(B2, HH, N0 + 16, k0, lane);
    acc1[0][0] = WMMA_BF16(a0, p0, acc1[0][0]);
    acc1[0][1] = WMMA_BF16(a0, p1, acc1[0][1]);
    acc1[1][0] = WMMA_BF16(a1, p0, acc1[1][0]);
    acc1[1][1] = WMMA_BF16(a1, p1, acc1[1][1]);
    acc2[0][0] = WMMA_BF16(a0, q0, acc2[0][0]);
    acc2[0][1] = WMMA_BF16(a0, q1, acc2[0][1]);
    acc2[1][0] = WMMA_BF16(a1, q0, acc2[1][0]);
    acc2[1][1] = WMMA_BF16(a1, q1, acc2[1][1]);
  }
  __bf16* hb = hbuf + (size_t)e * CAP * FF;
#pragma unroll
  for (int mi = 0; mi < 2; ++mi) {
#pragma unroll
    for (int ni = 0; ni < 2; ++ni) {
      int n = N0 + ni * 16 + (lane & 15);
      float bb1 = b1[(size_t)e * FF + n];
      float bb2 = b2[(size_t)e * FF + n];
#pragma unroll
      for (int r = 0; r < 8; ++r) {
        int m = M0 + mi * 16 + r + 8 * (lane >> 4);
        float v1 = acc1[mi][ni][r] + bb1, v2 = acc2[mi][ni][r] + bb2;
        float hv = (v1 / (1.0f + __expf(-v1))) * v2;   // silu(v1) * v2
        hb[(size_t)m * FF + n] = (__bf16)hv;
      }
    }
  }
}

// ---------------- 6) GEMM2 fused combine: out[tok] = gate * (h@w3 + b3) ----------------
// Block tile 128(M) x 64(N), 32x32 per wave.
__global__ __launch_bounds__(256) void moe_gemm2_kernel(const __bf16* __restrict__ hbuf,
                                                        const __bf16* __restrict__ w3t,
                                                        const float* __restrict__ b3,
                                                        const int* __restrict__ tok,
                                                        const float* __restrict__ gate,
                                                        float* __restrict__ out,
                                                        const int* __restrict__ rowlimit) {
  int e = blockIdx.z;
  int rl = rowlimit[e];
  if ((int)blockIdx.y * 128 >= rl) return;
  int wave = threadIdx.x >> 5, lane = threadIdx.x & 31;
  int wm = wave >> 1, wn = wave & 1;
  int M0 = blockIdx.y * 128 + wm * 32;
  int N0 = blockIdx.x * 64 + wn * 32;
  const __bf16* A = hbuf + (size_t)e * CAP * FF;
  const __bf16* B = w3t  + (size_t)e * HH * FF;
  v8f acc[2][2];
#pragma unroll
  for (int mi = 0; mi < 2; ++mi)
#pragma unroll
    for (int ni = 0; ni < 2; ++ni)
      acc[mi][ni] = {0.f,0.f,0.f,0.f,0.f,0.f,0.f,0.f};
  for (int k0 = 0; k0 < FF; k0 += 32) {
    v16bf a0 = load_a_tile(A, FF, M0,      k0, lane);
    v16bf a1 = load_a_tile(A, FF, M0 + 16, k0, lane);
    v16bf b0 = load_b_tile(B, FF, N0,      k0, lane);
    v16bf b1v = load_b_tile(B, FF, N0 + 16, k0, lane);
    acc[0][0] = WMMA_BF16(a0, b0,  acc[0][0]);
    acc[0][1] = WMMA_BF16(a0, b1v, acc[0][1]);
    acc[1][0] = WMMA_BF16(a1, b0,  acc[1][0]);
    acc[1][1] = WMMA_BF16(a1, b1v, acc[1][1]);
  }
  const int* tk = tok + (size_t)e * CAP;
#pragma unroll
  for (int mi = 0; mi < 2; ++mi) {
#pragma unroll
    for (int ni = 0; ni < 2; ++ni) {
      int n = N0 + ni * 16 + (lane & 15);
      float bb3 = b3[(size_t)e * HH + n];
#pragma unroll
      for (int r = 0; r < 8; ++r) {
        int m = M0 + mi * 16 + r + 8 * (lane >> 4);
        if (m < rl) {
          int t = tk[m];
          out[(size_t)t * HH + n] = gate[t] * (acc[mi][ni][r] + bb3);
        }
      }
    }
  }
}

// ---------------- 7) finalize aux (sequential -> deterministic) ----------------
__global__ void moe_finalize_kernel(const float* __restrict__ partials, int nblk,
                                    const int* __restrict__ rowlimit,
                                    float* __restrict__ aux_out) {
  if (blockIdx.x != 0 || threadIdx.x != 0) return;
  float imp[EE]; float z2 = 0.0f;
#pragma unroll
  for (int e = 0; e < EE; ++e) imp[e] = 0.0f;
  for (int b = 0; b < nblk; ++b) {
    const float* p = partials + (size_t)b * (EE + 1);
#pragma unroll
    for (int e = 0; e < EE; ++e) imp[e] += p[e];
    z2 += p[EE];
  }
  float a = 0.0f;
  const float invS = 1.0f / (float)SS;
#pragma unroll
  for (int e = 0; e < EE; ++e)
    a += (imp[e] * invS) * ((float)rowlimit[e] * invS);
  a = a * (float)EE * AUX_W + (z2 * invS) * Z_W;
  aux_out[0] = a;
}

// ---------------- host launch ----------------
extern "C" void kernel_launch(void* const* d_in, const int* in_sizes, int n_in,
                              void* d_out, int out_size, void* d_ws, size_t ws_size,
                              hipStream_t stream) {
  const float* x   = (const float*)d_in[0];
  const float* rw  = (const float*)d_in[1];
  const float* rb  = (const float*)d_in[2];
  const float* w1  = (const float*)d_in[3];
  const float* b1  = (const float*)d_in[4];
  const float* w2  = (const float*)d_in[5];
  const float* b2  = (const float*)d_in[6];
  const float* w3  = (const float*)d_in[7];
  const float* b3  = (const float*)d_in[8];
  float* out = (float*)d_out;

  // workspace carve
  uintptr_t p = (uintptr_t)d_ws;
  auto carve = [&](size_t bytes) -> void* {
    void* r = (void*)p;
    p += (bytes + 255) & ~(size_t)255;
    return r;
  };
  const size_t EHF = (size_t)EE * HH * FF;
  __bf16* w1t = (__bf16*)carve(EHF * sizeof(__bf16));
  __bf16* w2t = (__bf16*)carve(EHF * sizeof(__bf16));
  __bf16* w3t = (__bf16*)carve(EHF * sizeof(__bf16));
  __bf16* Xd  = (__bf16*)carve((size_t)EE * CAP * HH * sizeof(__bf16));
  __bf16* hbf = (__bf16*)carve((size_t)EE * CAP * FF * sizeof(__bf16));
  int* eidx   = (int*)carve((size_t)SS * sizeof(int));
  int* slot   = (int*)carve((size_t)SS * sizeof(int));
  float* gate = (float*)carve((size_t)SS * sizeof(float));
  int* tok    = (int*)carve((size_t)EE * CAP * sizeof(int));
  int* count  = (int*)carve(EE * sizeof(int));
  int* rlim   = (int*)carve(EE * sizeof(int));
  const int nRouterBlk = SS / 8;  // 2048
  float* partials = (float*)carve((size_t)nRouterBlk * (EE + 1) * sizeof(float));

  // 0) zero output (incl. aux slot; finalize overwrites it)
  moe_zero_kernel<<<4096, 256, 0, stream>>>(out, (size_t)SS * HH + 1);

  // 1) weight transpose+cast: w1,w2 [H][F]->[F][H]; w3 [F][H]->[H][F]
  dim3 tb(32, 8, 1);
  moe_transpose_kernel<<<dim3(FF / 32, HH / 32, EE), tb, 0, stream>>>(w1, w1t, HH, FF);
  moe_transpose_kernel<<<dim3(FF / 32, HH / 32, EE), tb, 0, stream>>>(w2, w2t, HH, FF);
  moe_transpose_kernel<<<dim3(HH / 32, FF / 32, EE), tb, 0, stream>>>(w3, w3t, FF, HH);

  // 2) router
  moe_router_kernel<<<nRouterBlk, 256, 0, stream>>>(x, rw, rb, eidx, gate, partials);

  // 3) capacity scan (exact reference token-order semantics)
  moe_scan_kernel<<<1, 256, 0, stream>>>(eidx, slot, count, rlim);

  // 4) gather into dispatch buffer
  moe_gather_kernel<<<SS / 8, 256, 0, stream>>>(x, eidx, slot, Xd, tok);

  // 5) expert FFN up-projection (fused SiLU-gate), bf16 WMMA, 128x64 block tiles
  moe_gemm1_kernel<<<dim3(FF / 64, CAP / 128, EE), 256, 0, stream>>>(Xd, w1t, w2t, b1, b2, hbf, rlim);

  // 6) down-projection fused with top-1 combine scatter, 128x64 block tiles
  moe_gemm2_kernel<<<dim3(HH / 64, CAP / 128, EE), 256, 0, stream>>>(hbf, w3t, b3, tok, gate, out, rlim);

  // 7) aux loss
  moe_finalize_kernel<<<1, 32, 0, stream>>>(partials, nRouterBlk, rlim, out + (size_t)SS * HH);
}
